// GroupedQueryAttention_71751723647807
// MI455X (gfx1250) — compile-verified, matrix operand order checked by reading, not run on hardware
//
#include <hip/hip_runtime.h>
#include <stdint.h>

#define D_MODEL 2048
#define KV_DIM  512
#define NUM_HEADS 32
#define NUM_KV_HEADS 8
#define D_K 64
#define GROUPS 4
#define BATCH 2
#define SEQ 2048
#define MROWS (BATCH * SEQ)

typedef _Float16 v16h __attribute__((ext_vector_type(16)));
typedef float    v8f  __attribute__((ext_vector_type(8)));
typedef uint32_t u32x4 __attribute__((ext_vector_type(4)));
typedef int      i32x8 __attribute__((ext_vector_type(8)));
typedef int      i32x4 __attribute__((ext_vector_type(4)));

#if defined(__has_builtin)
#if __has_builtin(__builtin_amdgcn_tensor_load_to_lds)
#define HAVE_TDM 1
#endif
#endif

// ---------------------------------------------------------------------------
// Fragment loader for both A and B operands of V_WMMA_F32_16X16X32_F16.
// Memory is interpreted as [row][K] with given stride (halves).
//   A operand: row = M, K = reduction dim.
//   B operand: row = N, K = reduction dim (i.e. B^T, N-major).
// Per ISA 7.12.2 (16-bit A 16x32): lane L holds row L&15; half-elements map to
//   low lanes : e0..7 -> K0..7,   e8..15 -> K16..23
//   high lanes: e0..7 -> K8..15,  e8..15 -> K24..31
// ---------------------------------------------------------------------------
__device__ __forceinline__ v16h load_frag_rowK(const _Float16* base, int stride,
                                               int row0, int lane) {
  const int r   = row0 + (lane & 15);
  const int khi = (lane & 16) ? 8 : 0;
  const _Float16* p = base + (size_t)r * stride + khi;
  v16h f;
#pragma unroll
  for (int i = 0; i < 4; ++i) {
    f[2 * i + 0] = p[2 * i + 0];
    f[2 * i + 1] = p[2 * i + 1];
    f[2 * i + 8] = p[2 * i + 16];
    f[2 * i + 9] = p[2 * i + 17];
  }
  return f;
}

__device__ __forceinline__ v8f wmma_f16(v16h a, v16h b, v8f c) {
  return __builtin_amdgcn_wmma_f32_16x16x32_f16(false, a, false, b, (short)0, c,
                                                false, false);
}

#if HAVE_TDM
// ---------------------------------------------------------------------------
// TDM: async DMA of a 2D 16-bit tile [rows x 32 halves] into LDS with the D#
// pad feature inserting 1 DWORD after every 16 DWORDs -> LDS row stride of
// 34 halves, matching load_frag_rowK's padded layout.
// D# packing per CDNA5 ISA:
//   g0: [1:0]=count=1, [63:32]=lds_addr, [120:64]=global_addr, [127:126]=2
//   g1: [17:16]=data_size(2B), [20]=pad_en, [24:22]=pad_interval(3 => 16 DW),
//       [31:25]=pad_amount(0 => 1 DW), [79:48]=tensor_dim0=32,
//       [111:80]=tensor_dim1=rows, [127:112]=tile_dim0=32,
//       [143:128]=tile_dim1=rows, [207:160]=tensor_dim0_stride
// Groups 2/3 unused for 2D tiles (tile_dim2 = 0) -> zeros.
// This toolchain's builtin is the 6-arg form:
//   (u32x4 g0, i32x8 g1, i32x4 g2, i32x4 g3, i32x8 pad, i32 cpol)
// ---------------------------------------------------------------------------
__device__ __forceinline__ void tdm_load_tile32(uint32_t lds_off,
                                                const _Float16* gsrc,
                                                uint32_t rows,
                                                uint32_t rowstride_elems) {
  const uint64_t ga = (uint64_t)(uintptr_t)gsrc;
  u32x4 g0;
  g0[0] = 1u;                                           // count=1
  g0[1] = lds_off;                                      // lds_addr
  g0[2] = (uint32_t)ga;                                 // global_addr[31:0]
  g0[3] = (uint32_t)((ga >> 32) & 0x01FFFFFFu) | (2u << 30);  // addr[56:32],type=2
  i32x8 g1;
  g1[0] = (int)((1u << 16) | (1u << 20) | (3u << 22));  // 2B elems, pad 16DW+1DW
  g1[1] = (int)(32u << 16);                             // tensor_dim0 = 32
  g1[2] = (int)(rows << 16);                            // tensor_dim1 = rows
  g1[3] = (int)(32u << 16);                             // tile_dim0 = 32
  g1[4] = (int)(rows & 0xFFFFu);                        // tile_dim1 = rows
  g1[5] = (int)rowstride_elems;                         // dim0_stride low32
  g1[6] = 0;
  g1[7] = 0;
  i32x4 z4 = {0, 0, 0, 0};
  i32x8 z8 = {0, 0, 0, 0, 0, 0, 0, 0};
  __builtin_amdgcn_tensor_load_to_lds(g0, g1, z4, z4, z8, 0);
}

__device__ __forceinline__ void tdm_wait0() {
#if __has_builtin(__builtin_amdgcn_s_wait_tensorcnt)
  __builtin_amdgcn_s_wait_tensorcnt(0);
#else
  asm volatile("s_wait_tensorcnt 0x0" ::: "memory");
#endif
}
#endif  // HAVE_TDM

// ---------------------------------------------------------------------------
// fp32 -> fp16 conversion (grid-stride)
// ---------------------------------------------------------------------------
__global__ void f32_to_f16_kernel(const float* __restrict__ src,
                                  _Float16* __restrict__ dst, int n) {
  int i = blockIdx.x * blockDim.x + threadIdx.x;
  int stride = gridDim.x * blockDim.x;
  for (; i < n; i += stride) dst[i] = (_Float16)src[i];
}

// ---------------------------------------------------------------------------
// Tiled WMMA GEMM: C[M,N] = A[M,K](f16) * W[K,N](f16) + bias[N](f32)
// Block: 256 threads (8 waves). Tile 128(M) x 64(N), K-step 32.
// Waves laid out 4x2; each wave computes a 32x32 sub-tile (2x2 WMMA frags).
// A tile is DMA'd into LDS by the Tensor Data Mover (wave 0 issues, pad
// feature gives the 34-half stride); B tile transposed manually (TDM cannot
// transpose). LDS padded to 34 halves -> conflict-free fragment pair loads.
// ---------------------------------------------------------------------------
template <bool STORE_F32>
__global__ __launch_bounds__(256) void gemm_f16_wmma(
    const _Float16* __restrict__ A, const _Float16* __restrict__ W,
    const float* __restrict__ bias, _Float16* __restrict__ outH,
    float* __restrict__ outF, int M, int N, int K) {
  __shared__ _Float16 As[128][34];
  __shared__ _Float16 Bs[64][34];

  const int tid  = threadIdx.x;
  const int lane = tid & 31;
  const int wave = tid >> 5;
  const int wm   = (wave & 3) * 32;
  const int wn   = (wave >> 2) * 32;
  const int m0   = blockIdx.y * 128;
  const int n0   = blockIdx.x * 64;

  v8f c00 = {}, c01 = {}, c10 = {}, c11 = {};

  const int arow = tid >> 1;          // 0..127   (fallback A staging)
  const int acol = (tid & 1) * 16;    // 0 or 16
  const int bkk  = tid >> 3;          // 0..31
  const int bn   = (tid & 7) * 8;     // 0..56
  (void)arow; (void)acol;

#if HAVE_TDM
  const uint32_t as_lds = (uint32_t)(uintptr_t)(&As[0][0]);
#endif

  for (int k0 = 0; k0 < K; k0 += 32) {
    // ---- stage A tile ----
#if HAVE_TDM
    if (wave == 0) {
      tdm_load_tile32(as_lds, A + (size_t)m0 * K + k0, 128u, (uint32_t)K);
    }
#else
    {
      const _Float16* src = A + (size_t)(m0 + arow) * K + k0 + acol;
      float4 t0 = *(const float4*)(src);
      float4 t1 = *(const float4*)(src + 8);
      _Float16 loc[16];
      *(float4*)(loc)     = t0;
      *(float4*)(loc + 8) = t1;
#pragma unroll
      for (int j = 0; j < 16; ++j) As[arow][acol + j] = loc[j];
    }
#endif
    // ---- stage B tile transposed: Bs[n][kk] = W[k0+kk][n0+n] ----
    {
      const _Float16* src = W + (size_t)(k0 + bkk) * N + n0 + bn;
      float4 t0 = *(const float4*)(src);
      _Float16 loc[8];
      *(float4*)(loc) = t0;
#pragma unroll
      for (int j = 0; j < 8; ++j) Bs[bn + j][bkk] = loc[j];
    }
    // hint L2 for next K-step of W
    if (k0 + 32 < K) {
      __builtin_prefetch(W + (size_t)(k0 + 32 + bkk) * N + n0 + bn, 0, 1);
    }
#if HAVE_TDM
    if (wave == 0) tdm_wait0();  // A tile landed in LDS
#endif
    __syncthreads();

    v16h a0 = load_frag_rowK(&As[wm][0],      34, 0, lane);
    v16h a1 = load_frag_rowK(&As[wm + 16][0], 34, 0, lane);
    v16h b0 = load_frag_rowK(&Bs[wn][0],      34, 0, lane);
    v16h b1 = load_frag_rowK(&Bs[wn + 16][0], 34, 0, lane);

    c00 = wmma_f16(a0, b0, c00);
    c01 = wmma_f16(a0, b1, c01);
    c10 = wmma_f16(a1, b0, c10);
    c11 = wmma_f16(a1, b1, c11);
    __syncthreads();
  }

  // ---- epilogue: C layout -> lane = col (l&15), reg r = row r + 8*(l>=16) ----
  const int hi8 = (lane & 16) ? 8 : 0;
  const int cn  = lane & 15;
  const int col0 = n0 + wn + cn;
  const float bia0 = bias[col0];
  const float bia1 = bias[col0 + 16];
#pragma unroll
  for (int r = 0; r < 8; ++r) {
    const size_t row  = (size_t)(m0 + wm + r + hi8);
    const size_t row1 = row + 16;
    const float v00 = c00[r] + bia0;
    const float v01 = c01[r] + bia1;
    const float v10 = c10[r] + bia0;
    const float v11 = c11[r] + bia1;
    if (STORE_F32) {
      outF[row * N + col0]       = v00;
      outF[row * N + col0 + 16]  = v01;
      outF[row1 * N + col0]      = v10;
      outF[row1 * N + col0 + 16] = v11;
    } else {
      outH[row * N + col0]       = (_Float16)v00;
      outH[row * N + col0 + 16]  = (_Float16)v01;
      outH[row1 * N + col0]      = (_Float16)v10;
      outH[row1 * N + col0 + 16] = (_Float16)v11;
    }
  }
}

// ---------------------------------------------------------------------------
// GQA flash attention. Block = 128 threads = 4 waves, all on one (b, h).
// Wave w owns queries [q0, q0+16). Streams keys in steps of 32.
//  - S = Q K^T via WMMA: A = Q frags loaded straight from global (contiguous
//    d-pairs), B = K^T frags loaded straight from global (K stored [key][d]
//    row-major == N-major B layout).
//  - online softmax: C-layout rows live in regs, cols in lanes -> per-row
//    reduction = shfl_xor over 16-lane half-wave.
//  - P staged in LDS in A layout; V staged transposed [d][key] in LDS (shared
//    by all 4 waves of the block -> B layout), then 4 WMMAs for P*V.
// ---------------------------------------------------------------------------
__global__ __launch_bounds__(128) void gqa_attention_wmma(
    const _Float16* __restrict__ Q, const _Float16* __restrict__ Km,
    const _Float16* __restrict__ Vm, _Float16* __restrict__ O) {
  __shared__ _Float16 Vt[64][34];        // V^T tile: [d][key], shared by block
  __shared__ _Float16 Ps[4][16][34];     // per-wave P tile: [q][key]

  const int tid  = threadIdx.x;
  const int lane = tid & 31;
  const int wave = tid >> 5;
  const int qb   = blockIdx.x;   // 0..SEQ/64-1
  const int h    = blockIdx.y;   // 0..31
  const int b    = blockIdx.z;   // 0..1
  const int kvh  = h / GROUPS;   // GQA: 4 query heads share one KV head
  const int q0   = qb * 64 + wave * 16;

  const _Float16* Qbase = Q  + ((size_t)b * SEQ + q0) * D_MODEL + h * D_K;
  const _Float16* Kbase = Km + ((size_t)b * SEQ) * KV_DIM + kvh * D_K;
  const _Float16* Vbase = Vm + ((size_t)b * SEQ) * KV_DIM + kvh * D_K;

  // Q fragments (16 x 64 = two 16x32 A frags), kept in registers all along.
  const v16h qa0 = load_frag_rowK(Qbase,      D_MODEL, 0, lane);
  const v16h qa1 = load_frag_rowK(Qbase + 32, D_MODEL, 0, lane);

  v8f o0 = {}, o1 = {}, o2 = {}, o3 = {};
  float mrun[8], lrun[8];
#pragma unroll
  for (int r = 0; r < 8; ++r) { mrun[r] = -1e30f; lrun[r] = 0.0f; }

  const float scale = 0.125f;       // 1/sqrt(64)
  const int vkey = tid >> 2;        // 0..31
  const int vd0  = (tid & 3) * 16;  // 0,16,32,48

  for (int k0 = 0; k0 < SEQ; k0 += 32) {
    // ---- stage V^T tile cooperatively: Vt[d][key] = V[k0+key][d] ----
    {
      const _Float16* src = Vbase + (size_t)(k0 + vkey) * KV_DIM + vd0;
      float4 t0 = *(const float4*)(src);
      float4 t1 = *(const float4*)(src + 8);
      _Float16 loc[16];
      *(float4*)(loc)     = t0;
      *(float4*)(loc + 8) = t1;
#pragma unroll
      for (int j = 0; j < 16; ++j) Vt[vd0 + j][vkey] = loc[j];
    }

    // ---- scores: two 16x16 frags covering keys [k0,k0+16) and [k0+16,k0+32)
    v8f s0 = {}, s1 = {};
    {
      v16h kb00 = load_frag_rowK(Kbase + (size_t)k0 * KV_DIM,            KV_DIM, 0, lane);
      v16h kb01 = load_frag_rowK(Kbase + (size_t)k0 * KV_DIM + 32,       KV_DIM, 0, lane);
      s0 = wmma_f16(qa0, kb00, s0);
      s0 = wmma_f16(qa1, kb01, s0);
      v16h kb10 = load_frag_rowK(Kbase + (size_t)(k0 + 16) * KV_DIM,      KV_DIM, 0, lane);
      v16h kb11 = load_frag_rowK(Kbase + (size_t)(k0 + 16) * KV_DIM + 32, KV_DIM, 0, lane);
      s1 = wmma_f16(qa0, kb10, s1);
      s1 = wmma_f16(qa1, kb11, s1);
    }

    // ---- online softmax over the 32 new keys ----
    {
      const int hi8  = (lane & 16) ? 8 : 0;
      const int keyc = lane & 15;
      float corr[8];
#pragma unroll
      for (int r = 0; r < 8; ++r) {
        float a  = s0[r] * scale;
        float bb = s1[r] * scale;
        float m  = fmaxf(a, bb);
#pragma unroll
        for (int msk = 1; msk <= 8; msk <<= 1)
          m = fmaxf(m, __shfl_xor(m, msk, 32));
        const float mnew = fmaxf(mrun[r], m);
        const float cfac = __expf(mrun[r] - mnew);
        const float p0   = __expf(a - mnew);
        const float p1   = __expf(bb - mnew);
        float ts = p0 + p1;
#pragma unroll
        for (int msk = 1; msk <= 8; msk <<= 1)
          ts += __shfl_xor(ts, msk, 32);
        lrun[r] = lrun[r] * cfac + ts;
        mrun[r] = mnew;
        corr[r] = cfac;
        Ps[wave][r + hi8][keyc]      = (_Float16)p0;
        Ps[wave][r + hi8][keyc + 16] = (_Float16)p1;
      }
#pragma unroll
      for (int r = 0; r < 8; ++r) {
        o0[r] *= corr[r]; o1[r] *= corr[r];
        o2[r] *= corr[r]; o3[r] *= corr[r];
      }
    }
    __syncthreads();  // Vt + Ps visible

    // ---- O += P(16x32) * V(32x64): 4 WMMAs over the 4 d-subtiles ----
    {
      v16h pa  = load_frag_rowK(&Ps[wave][0][0], 34, 0, lane);
      v16h vb0 = load_frag_rowK(&Vt[0][0],  34, 0, lane);
      v16h vb1 = load_frag_rowK(&Vt[16][0], 34, 0, lane);
      v16h vb2 = load_frag_rowK(&Vt[32][0], 34, 0, lane);
      v16h vb3 = load_frag_rowK(&Vt[48][0], 34, 0, lane);
      o0 = wmma_f16(pa, vb0, o0);
      o1 = wmma_f16(pa, vb1, o1);
      o2 = wmma_f16(pa, vb2, o2);
      o3 = wmma_f16(pa, vb3, o3);
    }
    __syncthreads();  // before next iteration overwrites Vt/Ps
  }

  // ---- normalize and write attention output (f16, [B*S, D_MODEL]) ----
  const int hi8 = (lane & 16) ? 8 : 0;
  const int dn  = lane & 15;
  _Float16* Obase = O + ((size_t)b * SEQ + q0) * D_MODEL + h * D_K;
#pragma unroll
  for (int r = 0; r < 8; ++r) {
    const float inv = 1.0f / lrun[r];
    const size_t row = (size_t)(r + hi8) * D_MODEL;
    Obase[row + dn]      = (_Float16)(o0[r] * inv);
    Obase[row + dn + 16] = (_Float16)(o1[r] * inv);
    Obase[row + dn + 32] = (_Float16)(o2[r] * inv);
    Obase[row + dn + 48] = (_Float16)(o3[r] * inv);
  }
}

// ---------------------------------------------------------------------------
// Host launcher
// ---------------------------------------------------------------------------
extern "C" void kernel_launch(void* const* d_in, const int* in_sizes, int n_in,
                              void* d_out, int out_size, void* d_ws,
                              size_t ws_size, hipStream_t stream) {
  (void)in_sizes; (void)n_in; (void)out_size; (void)ws_size;

  const float* x  = (const float*)d_in[0];
  const float* Wq = (const float*)d_in[1];
  const float* bq = (const float*)d_in[2];
  const float* Wk = (const float*)d_in[3];
  const float* bk = (const float*)d_in[4];
  const float* Wv = (const float*)d_in[5];
  const float* bv = (const float*)d_in[6];
  const float* Wo = (const float*)d_in[7];
  const float* bo = (const float*)d_in[8];
  float* out = (float*)d_out;

  char* ws = (char*)d_ws;
  size_t off = 0;
  auto alloc_h = [&](size_t nelem) {
    _Float16* p = (_Float16*)(ws + off);
    off += nelem * sizeof(_Float16);
    return p;
  };
  _Float16* xh  = alloc_h((size_t)MROWS * D_MODEL);   // 16 MB
  _Float16* Wqh = alloc_h((size_t)D_MODEL * D_MODEL); //  8 MB
  _Float16* Wkh = alloc_h((size_t)D_MODEL * KV_DIM);  //  2 MB
  _Float16* Wvh = alloc_h((size_t)D_MODEL * KV_DIM);  //  2 MB
  _Float16* Woh = alloc_h((size_t)D_MODEL * D_MODEL); //  8 MB
  _Float16* Qh  = alloc_h((size_t)MROWS * D_MODEL);   // 16 MB
  _Float16* Kh  = alloc_h((size_t)MROWS * KV_DIM);    //  4 MB
  _Float16* Vh  = alloc_h((size_t)MROWS * KV_DIM);    //  4 MB
  _Float16* Oh  = alloc_h((size_t)MROWS * D_MODEL);   // 16 MB

  // 1) fp32 -> fp16 conversions
  f32_to_f16_kernel<<<1024, 256, 0, stream>>>(x,  xh,  MROWS * D_MODEL);
  f32_to_f16_kernel<<<1024, 256, 0, stream>>>(Wq, Wqh, D_MODEL * D_MODEL);
  f32_to_f16_kernel<<<512,  256, 0, stream>>>(Wk, Wkh, D_MODEL * KV_DIM);
  f32_to_f16_kernel<<<512,  256, 0, stream>>>(Wv, Wvh, D_MODEL * KV_DIM);
  f32_to_f16_kernel<<<1024, 256, 0, stream>>>(Wo, Woh, D_MODEL * D_MODEL);

  // 2) Q/K/V projections (f16 outputs)
  gemm_f16_wmma<false><<<dim3(D_MODEL / 64, MROWS / 128), 256, 0, stream>>>(
      xh, Wqh, bq, Qh, nullptr, MROWS, D_MODEL, D_MODEL);
  gemm_f16_wmma<false><<<dim3(KV_DIM / 64, MROWS / 128), 256, 0, stream>>>(
      xh, Wkh, bk, Kh, nullptr, MROWS, KV_DIM, D_MODEL);
  gemm_f16_wmma<false><<<dim3(KV_DIM / 64, MROWS / 128), 256, 0, stream>>>(
      xh, Wvh, bv, Vh, nullptr, MROWS, KV_DIM, D_MODEL);

  // 3) GQA flash attention
  gqa_attention_wmma<<<dim3(SEQ / 64, NUM_HEADS, BATCH), 128, 0, stream>>>(
      Qh, Kh, Vh, Oh);

  // 4) output projection (f32 output + bias)
  gemm_f16_wmma<true><<<dim3(D_MODEL / 64, MROWS / 128), 256, 0, stream>>>(
      Oh, Woh, bo, nullptr, out, MROWS, D_MODEL, D_MODEL);
}
